// GCNClassifier_84782654423394
// MI455X (gfx1250) — matrix operand order repaired and checked by reading, not compiled, not run on hardware
//
#include <hip/hip_runtime.h>

// ---------------------------------------------------------------------------
// GraphSAGE (3x SAGEConv mean + BN + LeakyReLU) -> AvgPool -> MLP head
// CDNA5 / gfx1250: bf16 WMMA GEMMs (v_wmma_f32_16x16x32_bf16), wave32,
// double-buffered async global->LDS staging (GLOBAL_LOAD_ASYNC_TO_LDS_B128
// + s_wait_asynccnt) overlapping tile fill with WMMA compute.
// ---------------------------------------------------------------------------

#define DEVI __device__ __forceinline__

typedef __attribute__((ext_vector_type(16))) __bf16 v16bf;
typedef __attribute__((ext_vector_type(8)))  float  v8f;
typedef __attribute__((ext_vector_type(4)))  int    v4i;

#if defined(__has_builtin)
#if __has_builtin(__builtin_amdgcn_global_load_async_to_lds_b128)
#define HAS_ASYNC_LDS 1
#endif
#endif
#ifndef HAS_ASYNC_LDS
#define HAS_ASYNC_LDS 0
#endif

// builtin signature (verified via hipcc): (int4 AS1*, int4 AS3*, i32, i32)
typedef __attribute__((address_space(1))) v4i gv4i_t;
typedef __attribute__((address_space(3))) v4i lv4i_t;

DEVI void wait_async0() {
#if defined(__has_builtin) && __has_builtin(__builtin_amdgcn_s_wait_asynccnt)
  __builtin_amdgcn_s_wait_asynccnt(0);
#else
  asm volatile("s_wait_asynccnt 0" ::: "memory");
#endif
}

constexpr int N_NODES = 16384;
constexpr int E_EDGES = 65536;
constexpr int G_GRAPHS = 64;
constexpr int IN_F = 63;
constexpr int HID = 2048;
constexpr int MID = 1024;
constexpr int NCLS = 18;
constexpr float EPS = 1e-5f;
constexpr float SLOPE = 0.01f;

// f32 -> bf16 round-to-nearest-even (storage only)
DEVI unsigned short f2bf(float f) {
  unsigned u = __float_as_uint(f);
  unsigned r = u + 0x7FFFu + ((u >> 16) & 1u);
  return (unsigned short)(r >> 16);
}
DEVI float bf2f(unsigned short s) { return __uint_as_float(((unsigned)s) << 16); }

// ---------------------------------------------------------------------------
// Elementwise / utility kernels
// ---------------------------------------------------------------------------
__global__ void k_zero_f32(float* p, int n) {
  int i = blockIdx.x * 256 + threadIdx.x;
  if (i < n) p[i] = 0.f;
}

__global__ void k_cvt_bf16(const float* __restrict__ s, unsigned short* __restrict__ d, int n) {
  int i = blockIdx.x * 256 + threadIdx.x;
  if (i < n) d[i] = f2bf(s[i]);
}

// h[N,63] f32 -> xb[N,2048] bf16 (cols 0..62 data, col 63 zero pad)
__global__ void k_pad_x0(const float* __restrict__ h, unsigned short* __restrict__ xb) {
  int node = blockIdx.x;
  int c = threadIdx.x;                 // 0..63
  xb[node * HID + c] = (c < IN_F) ? f2bf(h[node * IN_F + c]) : (unsigned short)0;
}

// W[63,HID] f32 -> Wb[64,HID] bf16 (row 63 zeroed)
__global__ void k_pad_w1(const float* __restrict__ w, unsigned short* __restrict__ wb) {
  int idx = blockIdx.x * 256 + threadIdx.x;
  if (idx >= 64 * HID) return;
  int r = idx / HID, c = idx % HID;
  wb[idx] = (r < IN_F) ? f2bf(w[r * HID + c]) : (unsigned short)0;
}

__global__ void k_deg(const int* __restrict__ dst, float* __restrict__ deg) {
  int e = blockIdx.x * 256 + threadIdx.x;
  if (e < E_EDGES) unsafeAtomicAdd(&deg[dst[e]], 1.f);
}

// nsum[dst,:K] += x[src,:K]  (x bf16, row stride ldx)
__global__ void k_scatter(const unsigned short* __restrict__ x, int ldx,
                          const int* __restrict__ src, const int* __restrict__ dst,
                          float* __restrict__ nsum, int K) {
  int e = blockIdx.x;
  int f = blockIdx.y * blockDim.x + threadIdx.x;
  if (f >= K) return;
  int s = src[e], d = dst[e];
  unsafeAtomicAdd(&nsum[(long long)d * K + f], bf2f(x[(long long)s * ldx + f]));
}

// mb[node, f] (stride HID) = bf16( nsum[node*K+f] / max(deg,1) )
__global__ void k_mean(const float* __restrict__ nsum, const float* __restrict__ deg,
                       unsigned short* __restrict__ mb, int K) {
  int i = blockIdx.x * 256 + threadIdx.x;
  if (i >= N_NODES * K) return;
  int node = i / K, f = i - node * K;
  float dg = deg[node];
  dg = dg > 1.f ? dg : 1.f;
  mb[(long long)node * HID + f] = f2bf(nsum[i] / dg);
}

// ---------------------------------------------------------------------------
// WMMA GEMM:  C[M,Nc] = [A0 | A1] @ [B0 ; B1] + bias
//   A0,A1: bf16 [M, Khalf], row stride lda ; B0,B1: bf16 [Khalf, Nc], stride ldb
// Block: 256 threads (8 wave32). Tile BM=128 x BN=128, K-step 32.
// Wave grid 4(M) x 2(N): each wave = 32x64 C sub-tile = 2x4 WMMA accumulators.
// Double-buffered pipeline: async A(k+1)->LDS + B(k+1)->regs overlap WMMA(k);
// one workgroup barrier per K-step.
// ---------------------------------------------------------------------------
template <bool GUARD>
__global__ void __launch_bounds__(256) k_gemm_bf16(
    const unsigned short* __restrict__ A0, const unsigned short* __restrict__ A1,
    const unsigned short* __restrict__ B0, const unsigned short* __restrict__ B1,
    const float* __restrict__ bias, float* __restrict__ C,
    int M, int Nc, int Khalf, int lda, int ldb) {
  constexpr int BM = 128, BN = 128, BK = 32;
  constexpr int LDAS = 40;  // halfs: 80B rows -> 16B-aligned segs, 20-bank stride
  constexpr int LDBS = 34;  // transposed B: Bs[n*LDBS + k], 17-bank stride

  __shared__ unsigned short As[2][BM * LDAS];
  __shared__ unsigned short Bs[2][BN * LDBS];

  const int tid = threadIdx.x;
  const int lane = tid & 31;
  const int wave = tid >> 5;
  const int waveM = wave >> 1;        // 0..3  -> 32-row strip
  const int waveN = wave & 1;         // 0..1  -> 64-col strip
  const int rowBase = blockIdx.y * BM;
  const int colBase = blockIdx.x * BN;

  const int nb0 = Khalf / BK;
  const int nkb = 2 * nb0;

  v8f acc[2][4];
#pragma unroll
  for (int i = 0; i < 2; ++i)
#pragma unroll
    for (int j = 0; j < 4; ++j) acc[i][j] = (v8f){};

  // ---- pipeline stage helpers ----
  // A tile (128x32): 512 x 16B segments, 2 per thread, async global->LDS.
  auto stageA = [&](int kb, int buf) {
    const unsigned short* A = (kb < nb0) ? A0 : A1;
    const int k0 = ((kb < nb0) ? kb : kb - nb0) * BK;
#pragma unroll
    for (int i = 0; i < 2; ++i) {
      int li = tid * 2 + i;         // 0..511
      int r = li >> 2;              // 0..127
      int seg = li & 3;             // 0..3
      int gr = rowBase + r;
      if (GUARD) gr = (gr < M) ? gr : 0;  // clamp: dup row, discarded on store
      const unsigned short* gp = A + (long long)gr * lda + k0 + seg * 8;
      unsigned short* lp = &As[buf][r * LDAS + seg * 8];
#if HAS_ASYNC_LDS
      __builtin_amdgcn_global_load_async_to_lds_b128((gv4i_t*)gp, (lv4i_t*)lp, 0, 0);
#else
      *(uint4*)lp = *(const uint4*)gp;
#endif
    }
  };
  // B tile (32x128): 2 x 16B per thread -> registers.
  auto loadB = [&](int kb, uint4* reg) {
    const unsigned short* B = (kb < nb0) ? B0 : B1;
    const int k0 = ((kb < nb0) ? kb : kb - nb0) * BK;
#pragma unroll
    for (int i = 0; i < 2; ++i) {
      int li = tid * 2 + i;          // 0..511
      int krow = li >> 4;            // 0..31
      int nseg = li & 15;            // 0..15
      reg[i] = *(const uint4*)(B + (long long)(k0 + krow) * ldb + colBase + nseg * 8);
    }
  };
  // commit B registers to LDS, transposed -> Bs[buf][n][k]
  auto storeB = [&](const uint4* reg, int buf) {
#pragma unroll
    for (int i = 0; i < 2; ++i) {
      int li = tid * 2 + i;
      int krow = li >> 4;
      int nseg = li & 15;
      const unsigned short* hv = (const unsigned short*)&reg[i];
#pragma unroll
      for (int hh = 0; hh < 8; ++hh)
        Bs[buf][(nseg * 8 + hh) * LDBS + krow] = hv[hh];
    }
  };

  // ---- prologue: fill buffer 0 ----
  {
    uint4 breg[2];
    stageA(0, 0);
    loadB(0, breg);
    storeB(breg, 0);
#if HAS_ASYNC_LDS
    wait_async0();
#endif
  }
  __syncthreads();

  // ---- main pipelined loop ----
  for (int kb = 0; kb < nkb; ++kb) {
    const int cur = kb & 1;
    const int nxt = cur ^ 1;
    const bool more = (kb + 1 < nkb);

    uint4 breg[2];
    if (more) {
      stageA(kb + 1, nxt);   // async fill of next A tile (safe: nxt idle since kb-1 barrier)
      loadB(kb + 1, breg);   // next B tile into registers; latency hidden by WMMAs below
    }
    if (kb + 2 < nkb) {
      // prefetch B two steps ahead into near caches (-> global_prefetch_b8)
      const int kp = kb + 2;
      const unsigned short* Bp = (kp < nb0) ? B0 : B1;
      const int kp0 = ((kp < nb0) ? kp : kp - nb0) * BK;
      __builtin_prefetch(Bp + (long long)(kp0 + (tid >> 4)) * ldb + colBase, 0, 3);
    }

    // ---- compute on buffer `cur`: fragments per ISA 7.12.2 (wave32, 16-bit) ----
    {
      const int hi = lane >> 4;
      const int mr0 = waveM * 32 + (lane & 15);
      union Frag { v16bf v; unsigned u[8]; };
      Frag a[2], b[4];
#pragma unroll
      for (int v = 0; v < 8; ++v) {
        // A 16x32: VGPR v holds K pair; lanes 16..31 offset K by 8
        int ka = (v & 3) * 2 + (v >> 2) * 16 + hi * 8;
        a[0].u[v] = *(const unsigned*)&As[cur][mr0 * LDAS + ka];
        a[1].u[v] = *(const unsigned*)&As[cur][(mr0 + 16) * LDAS + ka];
        // B 32x16: same K order as A (VGPR v holds K pair; lanes 16..31 offset K by 8)
        int kbv = (v & 3) * 2 + (v >> 2) * 16 + hi * 8;
#pragma unroll
        for (int j = 0; j < 4; ++j) {
          int nc = waveN * 64 + j * 16 + (lane & 15);
          b[j].u[v] = *(const unsigned*)&Bs[cur][nc * LDBS + kbv];
        }
      }
#pragma unroll
      for (int i = 0; i < 2; ++i)
#pragma unroll
        for (int j = 0; j < 4; ++j)
          acc[i][j] = __builtin_amdgcn_wmma_f32_16x16x32_bf16(
              false, a[i].v, false, b[j].v, (short)0, acc[i][j], false, false);
    }

    if (more) {
      storeB(breg, nxt);     // commit next B tile (nxt not read this iteration)
#if HAS_ASYNC_LDS
      wait_async0();         // next A tile landed (ASYNCcnt)
#endif
    }
    __syncthreads();         // single barrier per K-step
  }

  // ---- store C (+bias); C/D layout: VGPR r -> row hi*8+r, col lane&15 ----
  const int hi = lane >> 4;
  const int nloc = lane & 15;
#pragma unroll
  for (int i = 0; i < 2; ++i) {
#pragma unroll
    for (int j = 0; j < 4; ++j) {
      int n = colBase + waveN * 64 + j * 16 + nloc;
      float bv = bias[n];
#pragma unroll
      for (int r = 0; r < 8; ++r) {
        int m = rowBase + waveM * 32 + i * 16 + hi * 8 + r;
        if (!GUARD || m < M) C[(long long)m * Nc + n] = acc[i][j][r] + bv;
      }
    }
  }
}

// ---------------------------------------------------------------------------
// BatchNorm (training stats) + LeakyReLU
// ---------------------------------------------------------------------------
__global__ void k_colstats(const float* __restrict__ x, float* __restrict__ csum,
                           float* __restrict__ csq, int M, int Nc, int rowsPerBlock) {
  int col = blockIdx.x * 256 + threadIdx.x;
  int r0 = blockIdx.y * rowsPerBlock;
  float s = 0.f, q = 0.f;
  for (int r = r0; r < r0 + rowsPerBlock; ++r) {
    float v = x[(long long)r * Nc + col];
    s += v; q += v * v;
  }
  unsafeAtomicAdd(&csum[col], s);
  unsafeAtomicAdd(&csq[col], q);
}

__global__ void k_bn_lrelu(const float* __restrict__ x, const float* __restrict__ csum,
                           const float* __restrict__ csq, const float* __restrict__ gamma,
                           const float* __restrict__ beta, unsigned short* __restrict__ y,
                           int M, int Nc) {
  int i = blockIdx.x * 256 + threadIdx.x;
  if (i >= M * Nc) return;
  int col = i % Nc;
  float invM = 1.f / (float)M;
  float mu = csum[col] * invM;
  float var = csq[col] * invM - mu * mu;
  float v = (x[i] - mu) * rsqrtf(var + EPS) * gamma[col] + beta[col];
  y[i] = f2bf(v > 0.f ? v : SLOPE * v);
}

// ---------------------------------------------------------------------------
// Graph average pooling
// ---------------------------------------------------------------------------
__global__ void k_cnt(const int* __restrict__ gid, float* __restrict__ cnt) {
  int i = blockIdx.x * 256 + threadIdx.x;
  if (i < N_NODES) unsafeAtomicAdd(&cnt[gid[i]], 1.f);
}

__global__ void k_pool(const unsigned short* __restrict__ x, const int* __restrict__ gid,
                       float* __restrict__ pooled) {
  int i = blockIdx.x * 256 + threadIdx.x;
  if (i >= N_NODES * HID) return;
  int node = i / HID, c = i - node * HID;
  unsafeAtomicAdd(&pooled[(long long)gid[node] * HID + c], bf2f(x[i]));
}

__global__ void k_pooldiv(const float* __restrict__ pooled, const float* __restrict__ cnt,
                          unsigned short* __restrict__ pb) {
  int i = blockIdx.x * 256 + threadIdx.x;
  if (i >= G_GRAPHS * HID) return;
  int g = i / HID;
  float c = cnt[g];
  c = c > 1.f ? c : 1.f;
  pb[i] = f2bf(pooled[i] / c);
}

__global__ void k_lrelu_bf16(const float* __restrict__ x, unsigned short* __restrict__ y, int n) {
  int i = blockIdx.x * 256 + threadIdx.x;
  if (i >= n) return;
  float v = x[i];
  y[i] = f2bf(v > 0.f ? v : SLOPE * v);
}

// final tiny layer: [64,1024] @ [1024,18] + b
__global__ void k_fc3(const unsigned short* __restrict__ x, const float* __restrict__ w,
                      const float* __restrict__ b, float* __restrict__ out) {
  int i = blockIdx.x * 256 + threadIdx.x;
  if (i >= G_GRAPHS * NCLS) return;
  int g = i / NCLS, c = i - g * NCLS;
  float s = b[c];
  for (int k = 0; k < MID; ++k) s += bf2f(x[g * MID + k]) * w[k * NCLS + c];
  out[i] = s;
}

// ---------------------------------------------------------------------------
// Host-side launch
// ---------------------------------------------------------------------------
static inline int cdiv(int a, int b) { return (a + b - 1) / b; }

extern "C" void kernel_launch(void* const* d_in, const int* in_sizes, int n_in,
                              void* d_out, int out_size, void* d_ws, size_t ws_size,
                              hipStream_t stream) {
  (void)in_sizes; (void)n_in; (void)out_size; (void)ws_size;
  const float* h    = (const float*)d_in[0];
  const int*   src  = (const int*)d_in[1];
  const int*   dst  = (const int*)d_in[2];
  const int*   gid  = (const int*)d_in[3];
  const float* Ws1  = (const float*)d_in[4];
  const float* Wn1  = (const float*)d_in[5];
  const float* b1   = (const float*)d_in[6];
  const float* Ws2  = (const float*)d_in[7];
  const float* Wn2  = (const float*)d_in[8];
  const float* b2   = (const float*)d_in[9];
  const float* Ws3  = (const float*)d_in[10];
  const float* Wn3  = (const float*)d_in[11];
  const float* b3   = (const float*)d_in[12];
  const float* g1   = (const float*)d_in[13];
  const float* be1  = (const float*)d_in[14];
  const float* g2   = (const float*)d_in[15];
  const float* be2  = (const float*)d_in[16];
  const float* g3   = (const float*)d_in[17];
  const float* be3  = (const float*)d_in[18];
  const float* fc1w = (const float*)d_in[19];
  const float* fc1b = (const float*)d_in[20];
  const float* fc2w = (const float*)d_in[21];
  const float* fc2b = (const float*)d_in[22];
  const float* fc3w = (const float*)d_in[23];
  const float* fc3b = (const float*)d_in[24];

  // -------- carve workspace --------
  char* p = (char*)d_ws;
  auto carve = [&](size_t bytes) -> char* {
    char* r = p;
    p += (bytes + 255) & ~(size_t)255;
    return r;
  };
  unsigned short* xb  = (unsigned short*)carve((size_t)N_NODES * HID * 2);
  unsigned short* mb  = (unsigned short*)carve((size_t)N_NODES * HID * 2);
  float* nsum = (float*)carve((size_t)N_NODES * HID * 4);  // aliased as GEMM out
  float* outf = nsum;                                      // safe: sequential stream
  float* deg  = (float*)carve((size_t)N_NODES * 4);
  unsigned short* w1s = (unsigned short*)carve((size_t)64 * HID * 2);
  unsigned short* w1n = (unsigned short*)carve((size_t)64 * HID * 2);
  unsigned short* w2s = (unsigned short*)carve((size_t)HID * HID * 2);
  unsigned short* w2n = (unsigned short*)carve((size_t)HID * HID * 2);
  unsigned short* w3s = (unsigned short*)carve((size_t)HID * HID * 2);
  unsigned short* w3n = (unsigned short*)carve((size_t)HID * HID * 2);
  unsigned short* wf1 = (unsigned short*)carve((size_t)HID * HID * 2);
  unsigned short* wf2 = (unsigned short*)carve((size_t)HID * MID * 2);
  float* csum   = (float*)carve((size_t)HID * 4);
  float* csq    = (float*)carve((size_t)HID * 4);
  float* pooled = (float*)carve((size_t)G_GRAPHS * HID * 4);
  float* cnt    = (float*)carve((size_t)G_GRAPHS * 4);
  unsigned short* pooledb = (unsigned short*)carve((size_t)G_GRAPHS * HID * 2);
  float* outm1  = (float*)carve((size_t)G_GRAPHS * HID * 4);
  unsigned short* m1b = (unsigned short*)carve((size_t)G_GRAPHS * HID * 2);
  float* outm2  = (float*)carve((size_t)G_GRAPHS * MID * 4);
  unsigned short* m2b = (unsigned short*)carve((size_t)G_GRAPHS * MID * 2);

  // -------- weight + input conversion --------
  k_pad_w1<<<cdiv(64 * HID, 256), 256, 0, stream>>>(Ws1, w1s);
  k_pad_w1<<<cdiv(64 * HID, 256), 256, 0, stream>>>(Wn1, w1n);
  k_cvt_bf16<<<cdiv(HID * HID, 256), 256, 0, stream>>>(Ws2, w2s, HID * HID);
  k_cvt_bf16<<<cdiv(HID * HID, 256), 256, 0, stream>>>(Wn2, w2n, HID * HID);
  k_cvt_bf16<<<cdiv(HID * HID, 256), 256, 0, stream>>>(Ws3, w3s, HID * HID);
  k_cvt_bf16<<<cdiv(HID * HID, 256), 256, 0, stream>>>(Wn3, w3n, HID * HID);
  k_cvt_bf16<<<cdiv(HID * HID, 256), 256, 0, stream>>>(fc1w, wf1, HID * HID);
  k_cvt_bf16<<<cdiv(HID * MID, 256), 256, 0, stream>>>(fc2w, wf2, HID * MID);
  k_pad_x0<<<N_NODES, 64, 0, stream>>>(h, xb);

  // -------- degrees (shared across layers) --------
  k_zero_f32<<<cdiv(N_NODES, 256), 256, 0, stream>>>(deg, N_NODES);
  k_deg<<<cdiv(E_EDGES, 256), 256, 0, stream>>>(dst, deg);

  const unsigned short* BsArr[3] = {w1s, w2s, w3s};
  const unsigned short* BnArr[3] = {w1n, w2n, w3n};
  const float* biasArr[3] = {b1, b2, b3};
  const float* gamArr[3]  = {g1, g2, g3};
  const float* betArr[3]  = {be1, be2, be3};
  const int KinArr[3] = {64, HID, HID};

  for (int l = 0; l < 3; ++l) {
    const int K = KinArr[l];
    // mean aggregation
    k_zero_f32<<<cdiv(N_NODES * K, 256), 256, 0, stream>>>(nsum, N_NODES * K);
    {
      int blk = K < 256 ? K : 256;
      dim3 grid(E_EDGES, cdiv(K, blk));
      k_scatter<<<grid, blk, 0, stream>>>(xb, HID, src, dst, nsum, K);
    }
    k_mean<<<cdiv(N_NODES * K, 256), 256, 0, stream>>>(nsum, deg, mb, K);

    // fused: out = [x | mean] @ [Ws ; Wn] + b   (bf16 WMMA, unguarded M)
    k_gemm_bf16<false><<<dim3(HID / 128, N_NODES / 128), 256, 0, stream>>>(
        xb, mb, BsArr[l], BnArr[l], biasArr[l], outf, N_NODES, HID, K, HID, HID);

    // BN batch stats + leaky relu -> next bf16 activations
    k_zero_f32<<<cdiv(HID, 256), 256, 0, stream>>>(csum, HID);
    k_zero_f32<<<cdiv(HID, 256), 256, 0, stream>>>(csq, HID);
    k_colstats<<<dim3(HID / 256, 64), 256, 0, stream>>>(outf, csum, csq, N_NODES, HID,
                                                        N_NODES / 64);
    k_bn_lrelu<<<cdiv(N_NODES * HID, 256), 256, 0, stream>>>(outf, csum, csq, gamArr[l],
                                                             betArr[l], xb, N_NODES, HID);
  }

  // -------- average pooling over graphs --------
  k_zero_f32<<<cdiv(G_GRAPHS * HID, 256), 256, 0, stream>>>(pooled, G_GRAPHS * HID);
  k_zero_f32<<<1, 256, 0, stream>>>(cnt, G_GRAPHS);
  k_cnt<<<cdiv(N_NODES, 256), 256, 0, stream>>>(gid, cnt);
  k_pool<<<cdiv(N_NODES * HID, 256), 256, 0, stream>>>(xb, gid, pooled);
  k_pooldiv<<<cdiv(G_GRAPHS * HID, 256), 256, 0, stream>>>(pooled, cnt, pooledb);

  // -------- MLP head (M = 64 rows, guarded) --------
  k_gemm_bf16<true><<<dim3(HID / 128, 1), 256, 0, stream>>>(
      pooledb, pooledb + 1024, wf1, wf1 + (size_t)1024 * HID, fc1b, outm1,
      G_GRAPHS, HID, 1024, HID, HID);
  k_lrelu_bf16<<<cdiv(G_GRAPHS * HID, 256), 256, 0, stream>>>(outm1, m1b, G_GRAPHS * HID);

  k_gemm_bf16<true><<<dim3(MID / 128, 1), 256, 0, stream>>>(
      m1b, m1b + 1024, wf2, wf2 + (size_t)1024 * MID, fc2b, outm2,
      G_GRAPHS, MID, 1024, HID, MID);
  k_lrelu_bf16<<<cdiv(G_GRAPHS * MID, 256), 256, 0, stream>>>(outm2, m2b, G_GRAPHS * MID);

  k_fc3<<<cdiv(G_GRAPHS * NCLS, 256), 256, 0, stream>>>(m2b, fc3w, fc3b, (float*)d_out);
}
